// LocalWindowAttention_73400991088871
// MI455X (gfx1250) — matrix operand order repaired and checked by reading, not compiled, not run on hardware
//
#include <hip/hip_runtime.h>

#define L_SEQ 8192
#define BATCH 4
#define CDIM  1024
#define NHEAD 16
#define HDIM  64
#define WIN   64
#define NWIN  (L_SEQ / WIN)       // 128
#define MROWS (BATCH * L_SEQ)     // 32768
#define NQ    (CDIM + NHEAD)      // 1040

typedef __attribute__((ext_vector_type(16))) __bf16 bf16x16;
typedef __attribute__((ext_vector_type(8)))  __bf16 bf16x8;
typedef __attribute__((ext_vector_type(8)))  float  f32x8;
typedef __attribute__((ext_vector_type(4)))  float  f32x4;

union FragAB { bf16x16 v; bf16x8 h[2]; };

static __device__ __forceinline__ f32x8 wmma_bf16(bf16x16 a, bf16x16 b, f32x8 c) {
  // D = A(16x32 bf16) * B(32x16 bf16) + C(16x16 f32)
  return __builtin_amdgcn_wmma_f32_16x16x32_bf16(false, a, false, b, (short)0, c, false, false);
}

// Issue one async global->LDS 16B copy (+ a second at +16B via inst offset,
// which per the ISA applies to BOTH the LDS and global addresses).
static __device__ __forceinline__ void async_copy32(unsigned lds_addr, const void* gptr) {
  const unsigned long long ga = (unsigned long long)gptr;
  asm volatile("global_load_async_to_lds_b128 %0, %1, off"
               :: "v"(lds_addr), "v"(ga) : "memory");
  asm volatile("global_load_async_to_lds_b128 %0, %1, off offset:16"
               :: "v"(lds_addr), "v"(ga) : "memory");
}
static __device__ __forceinline__ void wait_async0() {
  asm volatile("s_wait_asynccnt 0x0" ::: "memory");
}

// ---------------------------------------------------------------------------
// Weight transpose + f32 -> bf16 convert:  W (CDIM x ncols) -> Wt (ncols x CDIM)
// ---------------------------------------------------------------------------
__global__ __launch_bounds__(256) void wt_bf16_kernel(const float* __restrict__ W,
                                                      __bf16* __restrict__ Wt,
                                                      int ncols) {
  __shared__ float tile[32][33];
  const int tx = threadIdx.x, ty = threadIdx.y;
  const int c0 = blockIdx.y * 32;   // row of W (K dimension)
  const int n0 = blockIdx.x * 32;   // col of W (N dimension)
#pragma unroll
  for (int i = 0; i < 32; i += 8) {
    const int col = n0 + tx;
    tile[ty + i][tx] = (col < ncols) ? W[(size_t)(c0 + ty + i) * ncols + col] : 0.0f;
  }
  __syncthreads();
#pragma unroll
  for (int i = 0; i < 32; i += 8) {
    const int n = n0 + ty + i;
    if (n < ncols) Wt[(size_t)n * CDIM + (c0 + tx)] = (__bf16)tile[tx][ty + i];
  }
}

// ---------------------------------------------------------------------------
// Double-buffered bf16 WMMA GEMM: Out(M x nvalid) = A(M x 1024) * Bt^T + bias
//   MODE 0: A = query f32 (L,B,C)-indexed; out q bf16 (m,c) + sigmoid gate
//   MODE 1: A = key   f32;                 out k bf16 (m,c)
//   MODE 2: A = value f32;                 out v bf16 transposed (b,h,d,l)
//   MODE 3: A = attn-out bf16 (m,c);       out f32 (l,b,c)
// B tiles (and A tiles in MODE 3) are staged with global_load_async_to_lds;
// f32 A tiles go through registers for the f32->bf16 convert.
// Block: 256 threads (8 waves as 2Mx4N), tile 128x128x32.
// ---------------------------------------------------------------------------
template <int MODE>
__global__ __launch_bounds__(256) void gemm_kernel(const void* __restrict__ Ain,
                                                   const __bf16* __restrict__ Bt,
                                                   const float* __restrict__ bias,
                                                   void* __restrict__ Out,
                                                   float* __restrict__ gate_out,
                                                   int nvalid) {
  __shared__ __align__(16) __bf16 As[2][128][40];
  __shared__ __align__(16) __bf16 Bs[2][128][40];

  const int tid   = threadIdx.x;
  const int lane  = tid & 31;
  const int wv    = tid >> 5;
  const int waveM = wv & 1;        // 0..1  -> 64-row slab
  const int waveN = wv >> 1;       // 0..3  -> 32-col slab
  const int lm    = lane & 15;
  const int kg    = lane >> 4;

  const int m0 = blockIdx.x * 128;
  const int n0 = blockIdx.y * 128;

  f32x8 fz = {0.f, 0.f, 0.f, 0.f, 0.f, 0.f, 0.f, 0.f};
  f32x8 acc[4][2];
#pragma unroll
  for (int i = 0; i < 4; ++i)
#pragma unroll
    for (int j = 0; j < 2; ++j) acc[i][j] = fz;

  // staging assignments: thread -> (row = tid/2, col-half = (tid&1)*16)
  const int ar = tid >> 1;
  const int ac = (tid & 1) * 16;

  const float*  xrow = nullptr;
  const __bf16* arow = nullptr;
  if (MODE < 3) {
    const int m = m0 + ar;
    const int l = m & (L_SEQ - 1);
    const int b = m >> 13;
    xrow = (const float*)Ain + ((size_t)l * BATCH + b) * CDIM;   // X[l][b][:]
  } else {
    arow = (const __bf16*)Ain + (size_t)(m0 + ar) * CDIM;
  }
  // Clamp out-of-range weight rows to a valid row: garbage lands only in LDS
  // columns whose outputs are masked by the n >= nvalid store guard.
  int brow_idx = n0 + ar;
  if (brow_idx >= nvalid) brow_idx = n0;
  const __bf16* brow = Bt + (size_t)brow_idx * CDIM;

  auto asyncB = [&](int buf, int kk) {
    async_copy32((unsigned)(uintptr_t)&Bs[buf][ar][ac], brow + kk + ac);
  };
  auto asyncA3 = [&](int buf, int kk) {
    async_copy32((unsigned)(uintptr_t)&As[buf][ar][ac], arow + kk + ac);
  };
  auto loadA = [&](int kk, f32x4* r) {
    r[0] = *(const f32x4*)(xrow + kk + ac);
    r[1] = *(const f32x4*)(xrow + kk + ac + 4);
    r[2] = *(const f32x4*)(xrow + kk + ac + 8);
    r[3] = *(const f32x4*)(xrow + kk + ac + 12);
  };
  auto cvtStoreA = [&](int buf, const f32x4* r) {
    bf16x8 t0, t1;
#pragma unroll
    for (int i = 0; i < 4; ++i) {
      t0[i]     = (__bf16)r[0][i];
      t0[4 + i] = (__bf16)r[1][i];
      t1[i]     = (__bf16)r[2][i];
      t1[4 + i] = (__bf16)r[3][i];
    }
    *(bf16x8*)&As[buf][ar][ac]     = t0;
    *(bf16x8*)&As[buf][ar][ac + 8] = t1;
  };

  // ---- prologue: fill buffer 0 ----
  if (MODE < 3) {
    f32x4 r[4];
    loadA(0, r);
    cvtStoreA(0, r);
  } else {
    asyncA3(0, 0);
  }
  asyncB(0, 0);
  wait_async0();
  __syncthreads();

  // ---- main loop: one barrier per K-step, async prefetch into other buffer --
  for (int kk = 0; kk < CDIM; kk += 32) {
    const int cur = (kk >> 5) & 1;
    const bool more = (kk + 32) < CDIM;
    f32x4 pr[4];
    if (more) {
      asyncB(cur ^ 1, kk + 32);
      if (MODE == 3) asyncA3(cur ^ 1, kk + 32);
      else           loadA(kk + 32, pr);
    }

    // fragments (A and B both use contiguous k = 16*kg + s) + 8 WMMAs
    FragAB a[4], bf[2];
#pragma unroll
    for (int mt = 0; mt < 4; ++mt) {
      const int r = waveM * 64 + mt * 16 + lm;
      a[mt].h[0] = *(const bf16x8*)&As[cur][r][kg * 16];
      a[mt].h[1] = *(const bf16x8*)&As[cur][r][kg * 16 + 8];
    }
#pragma unroll
    for (int nt = 0; nt < 2; ++nt) {
      const int r = waveN * 32 + nt * 16 + lm;
      bf[nt].h[0] = *(const bf16x8*)&Bs[cur][r][kg * 16];
      bf[nt].h[1] = *(const bf16x8*)&Bs[cur][r][kg * 16 + 8];
    }
#pragma unroll
    for (int mt = 0; mt < 4; ++mt)
#pragma unroll
      for (int nt = 0; nt < 2; ++nt)
        acc[mt][nt] = wmma_bf16(a[mt].v, bf[nt].v, acc[mt][nt]);

    if (more) {
      if (MODE < 3) cvtStoreA(cur ^ 1, pr);
      wait_async0();
      __syncthreads();
    }
  }

  // ---- epilogue ----
#pragma unroll
  for (int nt = 0; nt < 2; ++nt) {
    const int n = n0 + waveN * 32 + nt * 16 + lm;
    if (n >= nvalid) continue;
    const float bb = bias[n];
#pragma unroll
    for (int mt = 0; mt < 4; ++mt) {
      const int mbase = m0 + waveM * 64 + mt * 16 + kg * 8;
      if (MODE == 2) {
        // store V transposed per head: vt[(b*H + h)*64 + d][l], 8 contiguous l
        const int bidx  = mbase >> 13;
        const int lbase = mbase & (L_SEQ - 1);
        const int hh = n >> 6, dd = n & 63;
        __bf16* dst = (__bf16*)Out +
                      ((size_t)((bidx * NHEAD + hh) * HDIM + dd)) * L_SEQ + lbase;
        bf16x8 pk;
#pragma unroll
        for (int j = 0; j < 8; ++j) pk[j] = (__bf16)(acc[mt][nt][j] + bb);
        *(bf16x8*)dst = pk;
      } else {
#pragma unroll
        for (int j = 0; j < 8; ++j) {
          const int m = mbase + j;
          const float val = acc[mt][nt][j] + bb;
          if (MODE == 0) {
            if (n < CDIM) {
              ((__bf16*)Out)[(size_t)m * CDIM + n] = (__bf16)val;
            } else {
              const int hh = n - CDIM;
              const int bidx = m >> 13;
              const int l = m & (L_SEQ - 1);
              gate_out[((size_t)(bidx * NHEAD + hh)) * L_SEQ + l] =
                  1.0f / (1.0f + __expf(-val));
            }
          } else if (MODE == 1) {
            ((__bf16*)Out)[(size_t)m * CDIM + n] = (__bf16)val;
          } else {  // MODE == 3 -> final output (L,B,C) f32
            const int bidx = m >> 13;
            const int l = m & (L_SEQ - 1);
            ((float*)Out)[((size_t)l * BATCH + bidx) * CDIM + n] = val;
          }
        }
      }
    }
  }
}

// ---------------------------------------------------------------------------
// Windowed attention: one wave per (b, h, window).
// S = QK^T/8 -> softmax -> P ; O = P V ; out = O * g (bf16 to workspace).
// ---------------------------------------------------------------------------
__global__ __launch_bounds__(128) void attn_kernel(const __bf16* __restrict__ q,
                                                   const __bf16* __restrict__ k,
                                                   const __bf16* __restrict__ vt,
                                                   const float* __restrict__ g,
                                                   __bf16* __restrict__ ao) {
  __shared__ __align__(16) __bf16 Ps[4][64][72];  // per-wave P tile, padded stride

  const int lane = threadIdx.x & 31;
  const int wv   = threadIdx.x >> 5;
  const int item = blockIdx.x * 4 + wv;           // 0 .. B*H*NWIN-1
  const int b = item >> 11;                        // H*NWIN = 2048
  const int h = (item >> 7) & (NHEAD - 1);
  const int w = item & (NWIN - 1);
  const int lm = lane & 15, kg = lane >> 4;

  const __bf16* qp = q + ((size_t)(b * L_SEQ + w * WIN)) * CDIM + h * HDIM;
  const __bf16* kp = k + ((size_t)(b * L_SEQ + w * WIN)) * CDIM + h * HDIM;
  const __bf16* vp = vt + ((size_t)((b * NHEAD + h) * HDIM)) * L_SEQ + (size_t)w * WIN;
  const float*  gp = g + ((size_t)(b * NHEAD + h)) * L_SEQ + w * WIN;
  __bf16*       op = ao + ((size_t)(b * L_SEQ + w * WIN)) * CDIM + h * HDIM;

  f32x8 fz = {0.f, 0.f, 0.f, 0.f, 0.f, 0.f, 0.f, 0.f};
  f32x8 s[4][4];
#pragma unroll
  for (int i = 0; i < 4; ++i)
#pragma unroll
    for (int j = 0; j < 4; ++j) s[i][j] = fz;

  // ---- S = Q * K^T (K rows are exactly the B-fragment columns: no transpose) ----
#pragma unroll
  for (int kc = 0; kc < 2; ++kc) {
    const int ko = kc * 32;
    FragAB a[4], bb[4];
#pragma unroll
    for (int mt = 0; mt < 4; ++mt) {
      const __bf16* p = qp + (size_t)(mt * 16 + lm) * CDIM + ko;
      a[mt].h[0] = *(const bf16x8*)(p + kg * 16);
      a[mt].h[1] = *(const bf16x8*)(p + kg * 16 + 8);
    }
#pragma unroll
    for (int nt = 0; nt < 4; ++nt) {
      const __bf16* p = kp + (size_t)(nt * 16 + lm) * CDIM + ko + kg * 16;
      bb[nt].h[0] = *(const bf16x8*)p;
      bb[nt].h[1] = *(const bf16x8*)(p + 8);
    }
#pragma unroll
    for (int mt = 0; mt < 4; ++mt)
#pragma unroll
      for (int nt = 0; nt < 4; ++nt)
        s[mt][nt] = wmma_bf16(a[mt].v, bb[nt].v, s[mt][nt]);
  }

  // ---- softmax over the 64 columns (scale = 1/sqrt(64) = 0.125) ----
  const float scale = 0.125f;
#pragma unroll
  for (int mt = 0; mt < 4; ++mt) {
#pragma unroll
    for (int j = 0; j < 8; ++j) {
      float rmax = -3.4e38f;
#pragma unroll
      for (int nt = 0; nt < 4; ++nt) rmax = fmaxf(rmax, s[mt][nt][j]);
#pragma unroll
      for (int off = 1; off < 16; off <<= 1)
        rmax = fmaxf(rmax, __shfl_xor(rmax, off, 16));
      float rsum = 0.0f;
#pragma unroll
      for (int nt = 0; nt < 4; ++nt) {
        const float e = __expf((s[mt][nt][j] - rmax) * scale);
        s[mt][nt][j] = e;
        rsum += e;
      }
#pragma unroll
      for (int off = 1; off < 16; off <<= 1) rsum += __shfl_xor(rsum, off, 16);
      const float inv = 1.0f / rsum;
#pragma unroll
      for (int nt = 0; nt < 4; ++nt) s[mt][nt][j] *= inv;
    }
  }

  // ---- reshape P: C-layout -> A-layout via LDS ----
#pragma unroll
  for (int mt = 0; mt < 4; ++mt)
#pragma unroll
    for (int nt = 0; nt < 4; ++nt)
#pragma unroll
      for (int j = 0; j < 8; ++j)
        Ps[wv][mt * 16 + j + 8 * kg][nt * 16 + lm] = (__bf16)s[mt][nt][j];

  // ---- O = P * V  (V already stored transposed (d, l): contiguous B-frags) ----
  f32x8 o[4][4];
#pragma unroll
  for (int i = 0; i < 4; ++i)
#pragma unroll
    for (int j = 0; j < 4; ++j) o[i][j] = fz;

#pragma unroll
  for (int kc = 0; kc < 2; ++kc) {
    const int ko = kc * 32;
    FragAB a[4], bb[4];
#pragma unroll
    for (int mt = 0; mt < 4; ++mt) {
      a[mt].h[0] = *(const bf16x8*)&Ps[wv][mt * 16 + lm][ko + kg * 16];
      a[mt].h[1] = *(const bf16x8*)&Ps[wv][mt * 16 + lm][ko + kg * 16 + 8];
    }
#pragma unroll
    for (int nt = 0; nt < 4; ++nt) {
      const __bf16* p = vp + (size_t)(nt * 16 + lm) * L_SEQ + ko + kg * 16;
      bb[nt].h[0] = *(const bf16x8*)p;
      bb[nt].h[1] = *(const bf16x8*)(p + 8);
    }
#pragma unroll
    for (int mt = 0; mt < 4; ++mt)
#pragma unroll
      for (int nt = 0; nt < 4; ++nt)
        o[mt][nt] = wmma_bf16(a[mt].v, bb[nt].v, o[mt][nt]);
  }

  // ---- apply gate, store bf16 attn-out (m, c) ----
#pragma unroll
  for (int mt = 0; mt < 4; ++mt)
#pragma unroll
    for (int nt = 0; nt < 4; ++nt)
#pragma unroll
      for (int j = 0; j < 8; ++j) {
        const int m = mt * 16 + j + 8 * kg;
        const float val = o[mt][nt][j] * gp[m];
        op[(size_t)m * CDIM + nt * 16 + lm] = (__bf16)val;
      }
}

// ---------------------------------------------------------------------------
extern "C" void kernel_launch(void* const* d_in, const int* in_sizes, int n_in,
                              void* d_out, int out_size, void* d_ws, size_t ws_size,
                              hipStream_t stream) {
  (void)in_sizes; (void)n_in; (void)out_size; (void)ws_size;
  const float* query = (const float*)d_in[0];
  const float* key_  = (const float*)d_in[1];
  const float* value = (const float*)d_in[2];
  const float* Wq = (const float*)d_in[3];
  const float* bq = (const float*)d_in[4];
  const float* Wk = (const float*)d_in[5];
  const float* bk = (const float*)d_in[6];
  const float* Wv = (const float*)d_in[7];
  const float* bv = (const float*)d_in[8];
  const float* Wo = (const float*)d_in[9];
  const float* bo = (const float*)d_in[10];

  float* outp = (float*)d_out;
  float* gout = outp + (size_t)L_SEQ * BATCH * CDIM;   // g region of the output tuple

  char* ws = (char*)d_ws;
  size_t off = 0;
  __bf16* wqT = (__bf16*)(ws + off); off += (size_t)NQ * CDIM * 2;
  __bf16* wkT = (__bf16*)(ws + off); off += (size_t)CDIM * CDIM * 2;
  __bf16* wvT = (__bf16*)(ws + off); off += (size_t)CDIM * CDIM * 2;
  __bf16* woT = (__bf16*)(ws + off); off += (size_t)CDIM * CDIM * 2;
  __bf16* qws  = (__bf16*)(ws + off); off += (size_t)MROWS * CDIM * 2;
  __bf16* kws  = (__bf16*)(ws + off); off += (size_t)MROWS * CDIM * 2;
  __bf16* vtws = (__bf16*)(ws + off); off += (size_t)MROWS * CDIM * 2;
  __bf16* aows = (__bf16*)(ws + off); off += (size_t)MROWS * CDIM * 2;

  // 1) weights -> bf16, transposed (N x K)
  dim3 tb(32, 8);
  wt_bf16_kernel<<<dim3((NQ + 31) / 32, CDIM / 32), tb, 0, stream>>>(Wq, wqT, NQ);
  wt_bf16_kernel<<<dim3(CDIM / 32, CDIM / 32), tb, 0, stream>>>(Wk, wkT, CDIM);
  wt_bf16_kernel<<<dim3(CDIM / 32, CDIM / 32), tb, 0, stream>>>(Wv, wvT, CDIM);
  wt_bf16_kernel<<<dim3(CDIM / 32, CDIM / 32), tb, 0, stream>>>(Wo, woT, CDIM);

  // 2) projections (q also emits sigmoid gate -> gout; v stored transposed per head)
  gemm_kernel<0><<<dim3(MROWS / 128, (NQ + 127) / 128), 256, 0, stream>>>(
      query, wqT, bq, qws, gout, NQ);
  gemm_kernel<1><<<dim3(MROWS / 128, CDIM / 128), 256, 0, stream>>>(
      key_, wkT, bk, kws, nullptr, CDIM);
  gemm_kernel<2><<<dim3(MROWS / 128, CDIM / 128), 256, 0, stream>>>(
      value, wvT, bv, vtws, nullptr, CDIM);

  // 3) windowed attention + gating
  attn_kernel<<<(BATCH * NHEAD * NWIN) / 4, 128, 0, stream>>>(qws, kws, vtws, gout, aows);

  // 4) output projection, permute back to (L, B, C) f32
  gemm_kernel<3><<<dim3(MROWS / 128, CDIM / 128), 256, 0, stream>>>(
      aows, woT, bo, outp, nullptr, CDIM);
}
